// GlobalEmdModelVn_38783554683624
// MI455X (gfx1250) — matrix operand order repaired and checked by reading, not compiled, not run on hardware
//
#include <hip/hip_runtime.h>
#include <hip/hip_bf16.h>

typedef __attribute__((ext_vector_type(16))) __bf16 v16bf;
typedef __attribute__((ext_vector_type(8)))  float  v8f;

#define EPSV 1e-7f
#define NPTS 131072
#define NSEG 32

__device__ __forceinline__ unsigned short f2bf(float f) {
    unsigned int u = __float_as_uint(f);
    unsigned int r = u + 0x7FFFu + ((u >> 16) & 1u);   // round-to-nearest-even
    return (unsigned short)(r >> 16);
}

__device__ __forceinline__ unsigned int ordf(float f) {
    unsigned int u = __float_as_uint(f);
    return (u & 0x80000000u) ? ~u : (u | 0x80000000u); // monotonic float->uint
}

// ---- CDNA5 async global->LDS copy (ASYNCcnt path) --------------------------
__device__ __forceinline__ void async_copy_b128(unsigned lds_addr, const void* g) {
    asm volatile("global_load_async_to_lds_b128 %0, %1, off"
                 :: "v"(lds_addr), "v"(g) : "memory");
}
__device__ __forceinline__ void wait_async0() {
#if __has_builtin(__builtin_amdgcn_s_wait_asynccnt)
    __builtin_amdgcn_s_wait_asynccnt(0);
#else
    asm volatile("s_wait_asynccnt 0x0" ::: "memory");
#endif
}
__device__ __forceinline__ unsigned lds_lo32(const void* p) {
    return (unsigned)(unsigned long long)p;   // LDS aperture: low 32 bits = LDS addr
}

union FragAB { v16bf v; unsigned int u[8]; };
union FragC  { v8f v; float f[8]; };

// ---------------------------------------------------------------------------
__global__ void cvt_w_kernel(const float* __restrict__ src,
                             unsigned short* __restrict__ dst, int n) {
    int i = blockIdx.x * blockDim.x + threadIdx.x;
    if (i < n) dst[i] = f2bf(src[i]);
}

__global__ void init_keys_kernel(unsigned long long* __restrict__ keys, int n) {
    int i = blockIdx.x * blockDim.x + threadIdx.x;
    if (i < n) keys[i] = 0ull;   // ordered-float 0 == smallest
}

// ---------------------------------------------------------------------------
// Fused VN GEMM kernel.
// MODE 0: dual GEMM + VN leaky-relu (p = Wf x, d = Wd x)
// MODE 1: single GEMM (p = Wf x)
// MODE 2: VN leaky-relu with identity p (p = x, d = Wd x); Cin == Cout
// MODE 3: segment scatter-max: d = Wd x, dot = sum_i x_i d_i, atomicMax key
// CONCAT: channels [0,CIN/2) from X (per point), [CIN/2,CIN) from X2[seg]
// ---------------------------------------------------------------------------
template <int MODE, bool CONCAT, int CIN>
__global__ __launch_bounds__(256)
void vn_gemm_kernel(const float* __restrict__ X,
                    const float* __restrict__ X2,
                    const int* __restrict__ ids,
                    const unsigned short* __restrict__ Wf,
                    const unsigned short* __restrict__ Wd,
                    float* __restrict__ OUT,
                    unsigned long long* __restrict__ keys,
                    int Cout) {
    constexpr int CinA = CONCAT ? CIN / 2 : CIN;   // per-point channels
    constexpr int CRAW = 16 * CinA * 3;            // floats in raw per-point tile
    __shared__ __align__(16) unsigned short xs[3][16][CIN];  // bf16 [i][pt][c]
    __shared__ __align__(16) float raw[CRAW];                // staged f32 tile
    __shared__ __align__(16) float rawB[CONCAT ? (CIN / 2) * 3 : 4];

    const int tid  = threadIdx.x;
    const int lane = tid & 31;
    const int wave = tid >> 5;
    const int half = lane >> 4;
    const int p0   = blockIdx.x * 16;
    const int o0   = blockIdx.y * 128 + wave * 16;

    // ---- async-stage raw f32 tile into LDS (contiguous block) --------------
    {
        const float* src = X + (size_t)p0 * CinA * 3;
        for (int j = tid; j < CRAW / 4; j += 256)
            async_copy_b128(lds_lo32(&raw[j * 4]), src + j * 4);
        if (CONCAT) {
            int seg = ids[p0];                      // whole tile shares a segment
            const float* s2 = X2 + (size_t)seg * (CIN / 2) * 3;
            for (int j = tid; j < ((CIN / 2) * 3) / 4; j += 256)
                async_copy_b128(lds_lo32(&rawB[j * 4]), s2 + j * 4);
        }
        wait_async0();
    }
    if (tid == 0 && !CONCAT)
        __builtin_prefetch(&X[((size_t)(p0 + 16) % NPTS) * (size_t)CinA * 3], 0, 1);
    __syncthreads();

    // ---- convert f32 -> bf16, transpose into WMMA-friendly layout ----------
    for (int idx = tid; idx < 16 * CIN * 3; idx += 256) {
        int p   = idx / (CIN * 3);
        int rem = idx - p * (CIN * 3);
        int c   = rem / 3;
        int i   = rem - c * 3;
        float v;
        if (CONCAT) v = (c < CinA) ? raw[p * CinA * 3 + c * 3 + i]
                                   : rawB[(c - CinA) * 3 + i];
        else        v = raw[idx];
        xs[i][p][c] = f2bf(v);
    }
    __syncthreads();

    // ---- accumulators ------------------------------------------------------
    FragC accP[3], accD[3];
#pragma unroll
    for (int i = 0; i < 3; ++i) {
#pragma unroll
        for (int r = 0; r < 8; ++r) { accP[i].f[r] = 0.f; accD[i].f[r] = 0.f; }
    }

    const int aM = lane & 15;
    const int bN = lane & 15;

#pragma unroll
    for (int k0 = 0; k0 < CIN; k0 += 32) {
        FragAB af, ad, b[3];
        // A fragment (16x32 bf16): lanes0-15 K=0..7 & 16..23, lanes16-31 +8
#pragma unroll
        for (int v = 0; v < 8; ++v) {
            int kk = k0 + ((v & 3) * 2) + ((v >> 2) * 16) + half * 8;
            if (MODE == 0 || MODE == 1)
                af.u[v] = *(const unsigned int*)&Wf[(size_t)(o0 + aM) * CIN + kk];
            if (MODE != 1)
                ad.u[v] = *(const unsigned int*)&Wd[(size_t)(o0 + aM) * CIN + kk];
        }
        // B fragment (32x16 bf16): lanes0-15 K=0..7 & 16..23, lanes16-31 +8
#pragma unroll
        for (int i = 0; i < 3; ++i) {
#pragma unroll
            for (int v = 0; v < 8; ++v) {
                int kk = k0 + ((v & 3) * 2) + ((v >> 2) * 16) + half * 8;
                b[i].u[v] = *(const unsigned int*)&xs[i][bN][kk];
            }
        }
#pragma unroll
        for (int i = 0; i < 3; ++i) {
            if (MODE == 0 || MODE == 1)
                accP[i].v = __builtin_amdgcn_wmma_f32_16x16x32_bf16(
                    false, af.v, false, b[i].v, (short)0, accP[i].v, false, false);
            if (MODE != 1)
                accD[i].v = __builtin_amdgcn_wmma_f32_16x16x32_bf16(
                    false, ad.v, false, b[i].v, (short)0, accD[i].v, false, false);
        }
    }

    // ---- epilogue ----------------------------------------------------------
    const int pt = p0 + (lane & 15);
#pragma unroll
    for (int r = 0; r < 8; ++r) {
        const int oc = o0 + r + (half ? 8 : 0);
        if (MODE == 1) {
            float* o = OUT + ((size_t)pt * Cout + oc) * 3;
            o[0] = accP[0].f[r]; o[1] = accP[1].f[r]; o[2] = accP[2].f[r];
        } else if (MODE == 0) {
            float pa = accP[0].f[r], pb = accP[1].f[r], pc = accP[2].f[r];
            float da = accD[0].f[r], db = accD[1].f[r], dc = accD[2].f[r];
            float dot = pa * da + pb * db + pc * dc;
            float s   = dot / (da * da + db * db + dc * dc + EPSV);
            bool pos  = dot >= 0.f;
            float* o = OUT + ((size_t)pt * Cout + oc) * 3;
            o[0] = pos ? pa : pa - s * da;
            o[1] = pos ? pb : pb - s * db;
            o[2] = pos ? pc : pc - s * dc;
        } else if (MODE == 2) {
            const float* xi = X + ((size_t)pt * Cout + oc) * 3;
            float pa = xi[0], pb = xi[1], pc = xi[2];
            float da = accD[0].f[r], db = accD[1].f[r], dc = accD[2].f[r];
            float dot = pa * da + pb * db + pc * dc;
            float s   = dot / (da * da + db * db + dc * dc + EPSV);
            bool pos  = dot >= 0.f;
            float* o = OUT + ((size_t)pt * Cout + oc) * 3;
            o[0] = pos ? pa : pa - s * da;
            o[1] = pos ? pb : pb - s * db;
            o[2] = pos ? pc : pc - s * dc;
        } else { // MODE == 3
            const float* xi = X + ((size_t)pt * Cout + oc) * 3;
            float dot = xi[0] * accD[0].f[r] + xi[1] * accD[1].f[r]
                      + xi[2] * accD[2].f[r];
            unsigned long long key =
                ((unsigned long long)ordf(dot) << 32) |
                (unsigned long long)(0xFFFFFFFFu - (unsigned int)pt);
            atomicMax(&keys[(size_t)ids[pt] * Cout + oc], key);
        }
    }
}

// ---------------------------------------------------------------------------
__global__ void gather_kernel(const unsigned long long* __restrict__ keys,
                              const float* __restrict__ X,
                              float* __restrict__ gseg, int C, int n) {
    int i = blockIdx.x * blockDim.x + threadIdx.x;
    if (i >= n) return;
    int c = i % C;
    unsigned int pt = 0xFFFFFFFFu - (unsigned int)(keys[i] & 0xFFFFFFFFull);
    const float* src = X + ((size_t)pt * C + c) * 3;
    float* dst = gseg + (size_t)i * 3;
    dst[0] = src[0]; dst[1] = src[1]; dst[2] = src[2];
}

// ---------------------------------------------------------------------------
extern "C" void kernel_launch(void* const* d_in, const int* in_sizes, int n_in,
                              void* d_out, int out_size, void* d_ws, size_t ws_size,
                              hipStream_t stream) {
    const float* pos = (const float*)d_in[0];
    const int*   ids = (const int*)d_in[1];

    char* ws = (char*)d_ws;
    unsigned short* wb = (unsigned short*)ws;                 // bf16 weights
    const size_t oW1f = 0, oW1d = 16384, oW2f = 32768, oW2d = 49152,
                 oW3 = 65536, oWd1 = 81920, oWr3 = 98304,
                 oW4f = 114688, oW4d = 180224, oW5 = 245760,
                 oWd2 = 311296, oWr5 = 376832;

    unsigned long long* key1 = (unsigned long long*)(ws + 1048576);   // 32*128
    unsigned long long* key2 = (unsigned long long*)(ws + 1081344);   // 32*256
    float* gseg1  = (float*)(ws + 1146880);   // [32,128,3]
    float* gseg1b = (float*)(ws + 1196032);   // [32,128,3]
    float* gseg2  = (float*)(ws + 1245184);   // [32,256,3]
    float* R0     = (float*)(ws + 2097152);   // [N,256,3] region
    float* R1     = (float*)(ws + 404750336); // [N,256,3] region
    float* outp   = (float*)d_out;            // [32,256,3]

    struct { int in; size_t off; int n; } wl[12] = {
        {2, oW1f, 16384}, {3, oW1d, 16384}, {4, oW2f, 16384}, {5, oW2d, 16384},
        {6, oW3, 16384},  {7, oWd1, 16384}, {8, oWr3, 16384},
        {9, oW4f, 65536}, {10, oW4d, 65536}, {11, oW5, 65536},
        {12, oWd2, 65536}, {13, oWr5, 65536}};
    for (int i = 0; i < 12; ++i)
        cvt_w_kernel<<<(wl[i].n + 255) / 256, 256, 0, stream>>>(
            (const float*)d_in[wl[i].in], wb + wl[i].off, wl[i].n);

    init_keys_kernel<<<(NSEG * (128 + 256) + 255) / 256, 256, 0, stream>>>(
        key1, NSEG * (128 + 256));   // key1,key2 contiguous

    dim3 blk(256);
    dim3 gN1(NPTS / 16, 1), gN2(NPTS / 16, 2);
    dim3 gS1(NSEG / 16, 1), gS2(NSEG / 16, 2);

    // vn1: pos[N,128,3] -> R0
    vn_gemm_kernel<0, false, 128><<<gN1, blk, 0, stream>>>(
        pos, nullptr, ids, wb + oW1f, wb + oW1d, R0, nullptr, 128);
    // vn2: R0 -> R1
    vn_gemm_kernel<0, false, 128><<<gN1, blk, 0, stream>>>(
        R0, nullptr, ids, wb + oW2f, wb + oW2d, R1, nullptr, 128);
    // vn3: R1 -> R0
    vn_gemm_kernel<1, false, 128><<<gN1, blk, 0, stream>>>(
        R1, nullptr, ids, wb + oW3, nullptr, R0, nullptr, 128);
    // segmax pool 1 over R0
    vn_gemm_kernel<3, false, 128><<<gN1, blk, 0, stream>>>(
        R0, nullptr, ids, nullptr, wb + oWd1, nullptr, key1, 128);
    gather_kernel<<<(NSEG * 128 + 255) / 256, 256, 0, stream>>>(
        key1, R0, gseg1, 128, NSEG * 128);
    // relu3 on segments: gseg1 -> gseg1b
    vn_gemm_kernel<2, false, 128><<<gS1, blk, 0, stream>>>(
        gseg1, nullptr, ids, nullptr, wb + oWr3, gseg1b, nullptr, 128);
    // vn4 (concat pos + broadcast gseg1b): -> R1
    vn_gemm_kernel<0, true, 256><<<gN2, blk, 0, stream>>>(
        pos, gseg1b, ids, wb + oW4f, wb + oW4d, R1, nullptr, 256);
    // vn5: R1 -> R0
    vn_gemm_kernel<1, false, 256><<<gN2, blk, 0, stream>>>(
        R1, nullptr, ids, wb + oW5, nullptr, R0, nullptr, 256);
    // segmax pool 2 over R0
    vn_gemm_kernel<3, false, 256><<<gN2, blk, 0, stream>>>(
        R0, nullptr, ids, nullptr, wb + oWd2, nullptr, key2, 256);
    gather_kernel<<<(NSEG * 256 + 255) / 256, 256, 0, stream>>>(
        key2, R0, gseg2, 256, NSEG * 256);
    // relu5 on segments: gseg2 -> d_out
    vn_gemm_kernel<2, false, 256><<<gS2, blk, 0, stream>>>(
        gseg2, nullptr, ids, nullptr, wb + oWr5, outp, nullptr, 256);

    (void)in_sizes; (void)n_in; (void)out_size; (void)ws_size;
}
